// SchNetAuTopology_56023553409598
// MI455X (gfx1250) — matrix operand order repaired and checked by reading, not compiled, not run on hardware
//
#include <hip/hip_runtime.h>
#include <hip/hip_bf16.h>
#include <math.h>

// ---------------- problem constants (match reference) ----------------
#define NA_    30000
#define NE_    600000
#define NMOL_  128
#define NF_    128      // NAB == NF == 128
#define NGR_   50       // real gaussian count
#define NGP_   64       // padded to K multiple of 32
#define NCONV_ 3
#define CUTOFF_ 5.0f

typedef _Float16 half16 __attribute__((ext_vector_type(16)));
typedef _Float16 half8  __attribute__((ext_vector_type(8)));
typedef float    f32x8  __attribute__((ext_vector_type(8)));
typedef int      i32x4  __attribute__((vector_size(16)));   // matches builtin proto

#define GLOBAL_AS __attribute__((address_space(1)))
#define LDS_AS    __attribute__((address_space(3)))

#if defined(__gfx1250__) && __has_builtin(__builtin_amdgcn_global_load_async_to_lds_b128) && __has_builtin(__builtin_amdgcn_s_wait_asynccnt)
#define HAVE_ASYNC_LDS 1
#else
#define HAVE_ASYNC_LDS 0
#endif

__device__ __forceinline__ float ssp_f(float x) {
  // softplus(x) - ln 2, overflow-safe
  float sp = (x > 20.0f) ? x : log1pf(__expf(x));
  return sp - 0.69314718055994531f;
}

__device__ __forceinline__ half16 ld_h16(const _Float16* p) {
  half8 lo = *(const half8*)p;
  half8 hi = *(const half8*)(p + 8);
  half16 r;
  #pragma unroll
  for (int t = 0; t < 8; ++t) { r[t] = lo[t]; r[8 + t] = hi[t]; }
  return r;
}

__device__ __forceinline__ f32x8 wmma_f16(half16 a, half16 b, f32x8 c) {
  return __builtin_amdgcn_wmma_f32_16x16x32_f16(false, a, false, b,
                                                (short)0, c, false, false);
}

// ---------------- node-side WMMA GEMM (templated: fully unrolled) -----------
// out = act(A @ W + bias); A: [M,K] fp32, W: [K,N] fp16,
// out: [M,N] fp32. MODE: 0 = plain, 1 = shifted-softplus, 2 = out += (plain).
// 256 threads = 8 wave32; each wave computes a 16-row x N tile.
// Weights are staged once per block into bank-padded LDS via the gfx1250
// async global->LDS path (ASYNCcnt) when available.
template <int K, int N, int MODE>
__global__ __launch_bounds__(256) void k_gemm(const float* __restrict__ A,
                                              const _Float16* __restrict__ W,
                                              const float* __restrict__ bias,
                                              float* __restrict__ out, int M) {
  constexpr int SWS = N + 8;                       // padded LDS row stride
  __shared__ __align__(16) _Float16 sW[K * SWS];

  const int tid  = threadIdx.x;
  const int lane = tid & 31;
  const int wave = tid >> 5;
  const int m    = lane & 15;
  const int hi   = lane >> 4;

  // ---- stage weights K x N into LDS ----
#if HAVE_ASYNC_LDS
  {
    constexpr int CHUNKS = K * N / 8;              // 16-byte chunks
    #pragma unroll
    for (int c0 = 0; c0 < CHUNKS; c0 += 256) {
      const int c    = c0 + tid;
      const int rowk = c / (N / 8);
      const int col8 = c - rowk * (N / 8);
      __builtin_amdgcn_global_load_async_to_lds_b128(
          (GLOBAL_AS i32x4*)(W + (size_t)c * 8),
          (LDS_AS i32x4*)(sW + rowk * SWS + col8 * 8), 0, 0);
    }
    __builtin_amdgcn_s_wait_asynccnt(0);
  }
#else
  for (int idx = tid; idx < K * N; idx += 256) {
    int kk = idx / N, nn = idx - kk * N;
    sW[kk * SWS + nn] = W[idx];
  }
#endif
  __syncthreads();

  const int row0 = blockIdx.x * 128 + wave * 16;
  const int row  = row0 + m;
  const bool rowok = (row < M);
  const float* Arow = A + (size_t)(rowok ? row : 0) * K;

  constexpr int NT = N / 16;
  f32x8 acc[NT];
  #pragma unroll
  for (int nt = 0; nt < NT; ++nt)
    #pragma unroll
    for (int t = 0; t < 8; ++t) acc[nt][t] = 0.0f;

  #pragma unroll
  for (int ks = 0; ks < K / 32; ++ks) {
    const int k0 = ks * 32;
    // A fragment: lane -> M, VGPR v(0..3) -> K=2v+8*hi, v(4..7) -> K=16+2(v-4)+8*hi
    half16 a;
    #pragma unroll
    for (int t = 0; t < 8; ++t) {
      float v0 = rowok ? Arow[k0 + hi * 8 + t]      : 0.0f;
      float v1 = rowok ? Arow[k0 + 16 + hi * 8 + t] : 0.0f;
      a[t]     = (_Float16)v0;
      a[8 + t] = (_Float16)v1;
    }
    // B fragments: lane -> N = lane&15, element order over K matches A
    const _Float16* wrow = sW + (k0 + hi * 8) * SWS + m;
    #pragma unroll
    for (int nt = 0; nt < NT; ++nt) {
      half16 b;
      #pragma unroll
      for (int t = 0; t < 8; ++t) { b[t] = wrow[t * SWS + nt * 16]; b[8 + t] = wrow[(16 + t) * SWS + nt * 16]; }
      acc[nt] = wmma_f16(a, b, acc[nt]);
    }
  }

  // D layout: lane -> n = lane&15, VGPR v -> m = v + 8*hi
  #pragma unroll
  for (int nt = 0; nt < NT; ++nt) {
    const int n = nt * 16 + m;
    const float bn = bias[n];
    #pragma unroll
    for (int v = 0; v < 8; ++v) {
      const int mm = row0 + hi * 8 + v;
      if (mm < M) {
        float x = acc[nt][v] + bn;
        if (MODE == 1) x = ssp_f(x);
        if (MODE == 2) out[(size_t)mm * N + n] += x;
        else           out[(size_t)mm * N + n]  = x;
      }
    }
  }
}

// ---------------- fused edge pipeline -------------------------------------
// Per 16-edge wave tile:
//   g = gaussians(d)  (generated in-register, K=64 incl. zero padding)
//   t = ssp(g @ W1 + b1)           -> staged fp16 in LDS (D->A transpose)
//   filt = t @ W2 + b2
//   atomically scatter h[j]*filt into agg[i] and h[i]*filt into agg[j]
// Weights read from global (L0/L2-resident, 16 KB / 32 KB fp16 matrices).
__global__ __launch_bounds__(256) void k_edge_fused(
    const float* __restrict__ dists,
    const _Float16* __restrict__ W1, const float* __restrict__ b1,
    const _Float16* __restrict__ W2, const float* __restrict__ b2,
    const int* __restrict__ nbr,
    const float* __restrict__ h,
    float* __restrict__ agg, int M) {
  constexpr int ST = NF_ + 8;                       // padded LDS row stride
  __shared__ __align__(16) _Float16 sT[8 * 16 * ST];

  const int tid  = threadIdx.x;
  const int lane = tid & 31;
  const int wave = tid >> 5;
  const int m    = lane & 15;
  const int hi   = lane >> 4;
  const int row0 = blockIdx.x * 128 + wave * 16;
  const int row  = row0 + m;
  const bool rowok = (row < M);
  const float d  = rowok ? dists[row] : 0.0f;
  const float width = CUTOFF_ / (NGR_ - 1);
  const float coef  = -0.5f / (width * width);
  _Float16* myT = sT + wave * 16 * ST;

  // ---- stage 1: K = 64 (gaussians computed on the fly), N = 128 ----
  {
    f32x8 acc[8];
    #pragma unroll
    for (int nt = 0; nt < 8; ++nt)
      #pragma unroll
      for (int t = 0; t < 8; ++t) acc[nt][t] = 0.0f;

    #pragma unroll
    for (int ks = 0; ks < 2; ++ks) {
      const int k0 = ks * 32;
      half16 a;
      #pragma unroll
      for (int t = 0; t < 8; ++t) {
        const int ka = k0 + hi * 8 + t;
        const int kb = k0 + 16 + hi * 8 + t;
        float da = d - width * (float)ka;
        float db = d - width * (float)kb;
        float ga = (rowok && ka < NGR_) ? __expf(coef * da * da) : 0.0f;
        float gb = (rowok && kb < NGR_) ? __expf(coef * db * db) : 0.0f;
        a[t]     = (_Float16)ga;
        a[8 + t] = (_Float16)gb;
      }
      const _Float16* wrow = W1 + (size_t)(k0 + hi * 8) * NF_ + m;
      #pragma unroll
      for (int nt = 0; nt < 8; ++nt) {
        half16 b;
        #pragma unroll
        for (int t = 0; t < 8; ++t) { b[t] = wrow[t * NF_ + nt * 16]; b[8 + t] = wrow[(16 + t) * NF_ + nt * 16]; }
        acc[nt] = wmma_f16(a, b, acc[nt]);
      }
    }

    // ssp(acc + b1) -> LDS (D layout: n = nt*16 + m, row = v + 8*hi)
    #pragma unroll
    for (int nt = 0; nt < 8; ++nt) {
      const int n = nt * 16 + m;
      const float bn = b1[n];
      #pragma unroll
      for (int v = 0; v < 8; ++v)
        myT[(hi * 8 + v) * ST + n] = (_Float16)ssp_f(acc[nt][v] + bn);
    }
  }
  __syncthreads();

  // ---- stage 2: filt = t @ W2 + b2, K = 128, A from LDS ----
  f32x8 acc2[8];
  #pragma unroll
  for (int nt = 0; nt < 8; ++nt)
    #pragma unroll
    for (int t = 0; t < 8; ++t) acc2[nt][t] = 0.0f;

  const _Float16* trow = myT + m * ST;
  #pragma unroll
  for (int ks = 0; ks < 4; ++ks) {
    const int k0 = ks * 32;
    half16 a;
    half8 alo = *(const half8*)(trow + k0 + hi * 8);
    half8 ahi = *(const half8*)(trow + k0 + 16 + hi * 8);
    #pragma unroll
    for (int t = 0; t < 8; ++t) { a[t] = alo[t]; a[8 + t] = ahi[t]; }
    const _Float16* wrow = W2 + (size_t)(k0 + hi * 8) * NF_ + m;
    #pragma unroll
    for (int nt = 0; nt < 8; ++nt) {
      half16 b;
      #pragma unroll
      for (int t = 0; t < 8; ++t) { b[t] = wrow[t * NF_ + nt * 16]; b[8 + t] = wrow[(16 + t) * NF_ + nt * 16]; }
      acc2[nt] = wmma_f16(a, b, acc2[nt]);
    }
  }

  // ---- message epilogue: gather h, scatter-add into agg ----
  int ii[8], jj[8];
  #pragma unroll
  for (int v = 0; v < 8; ++v) {
    const int e = row0 + hi * 8 + v;
    const bool ok = (e < M);
    ii[v] = ok ? nbr[2 * e]     : 0;
    jj[v] = ok ? nbr[2 * e + 1] : 0;
  }
  #pragma unroll
  for (int nt = 0; nt < 8; ++nt) {
    const int n = nt * 16 + m;
    const float bn = b2[n];
    #pragma unroll
    for (int v = 0; v < 8; ++v) {
      const int e = row0 + hi * 8 + v;
      if (e < M) {
        const float fl = acc2[nt][v] + bn;
        const float hj = h[(size_t)jj[v] * NF_ + n];
        const float hi_ = h[(size_t)ii[v] * NF_ + n];
        unsafeAtomicAdd(&agg[(size_t)ii[v] * NF_ + n], hj * fl);
        unsafeAtomicAdd(&agg[(size_t)jj[v] * NF_ + n], hi_ * fl);
      }
    }
  }
}

// ---------------- elementwise / irregular kernels ----------------
__global__ void k_zero(float* __restrict__ p, long n) {
  long t = (long)blockIdx.x * blockDim.x + threadIdx.x;
  if (t < n) p[t] = 0.0f;
}

__global__ void k_cvt(const float* __restrict__ s, _Float16* __restrict__ d, long n) {
  long t = (long)blockIdx.x * blockDim.x + threadIdx.x;
  if (t < n) d[t] = (_Float16)s[t];
}

// edge_W1 [3,50,128] -> [3,64,128] fp16, zero-padded rows 50..63
__global__ void k_cvt_pad_w1(const float* __restrict__ s, _Float16* __restrict__ d) {
  long t = (long)blockIdx.x * blockDim.x + threadIdx.x;
  if (t >= 3L * NGP_ * NF_) return;
  int l   = (int)(t / (NGP_ * NF_));
  int rem = (int)(t - (long)l * NGP_ * NF_);
  int kk  = rem / NF_, nn = rem - kk * NF_;
  d[t] = (kk < NGR_) ? (_Float16)s[((long)l * NGR_ + kk) * NF_ + nn] : (_Float16)0.0f;
}

__global__ void k_dist(const float* __restrict__ xyz, const int* __restrict__ nbr,
                       float* __restrict__ dists, int ne) {
  int e = blockIdx.x * blockDim.x + threadIdx.x;
  if (e >= ne) return;
  int i = nbr[2 * e], j = nbr[2 * e + 1];
  float dx = xyz[3 * i]     - xyz[3 * j];
  float dy = xyz[3 * i + 1] - xyz[3 * j + 1];
  float dz = xyz[3 * i + 2] - xyz[3 * j + 2];
  dists[e] = sqrtf(dx * dx + dy * dy + dz * dz);
}

__global__ void k_embed(const int* __restrict__ z, const float* __restrict__ embed,
                        float* __restrict__ r, long n) {
  long t = (long)blockIdx.x * blockDim.x + threadIdx.x;
  if (t >= n) return;
  int a = (int)(t >> 7), f = (int)(t & 127);
  r[t] = embed[(long)z[a] * NF_ + f];
}

__global__ void k_readout2(const float* __restrict__ t64,
                           const float* __restrict__ w2,
                           const float* __restrict__ b2,
                           const int* __restrict__ mol,
                           float* __restrict__ Eout, int na) {
  int a = blockIdx.x * blockDim.x + threadIdx.x;
  if (a >= na) return;
  float acc = b2[0];
  const float* tr = t64 + (size_t)a * 64;
  #pragma unroll
  for (int c = 0; c < 64; ++c) acc += tr[c] * w2[c];
  unsafeAtomicAdd(&Eout[mol[a]], acc);
}

// ---------------- launcher ----------------
extern "C" void kernel_launch(void* const* d_in, const int* in_sizes, int n_in,
                              void* d_out, int out_size, void* d_ws, size_t ws_size,
                              hipStream_t stream) {
  (void)in_sizes; (void)n_in; (void)out_size; (void)ws_size;
  const int*   z       = (const int*)d_in[0];
  const float* xyz     = (const float*)d_in[1];
  const int*   nbr     = (const int*)d_in[2];
  const int*   mol     = (const int*)d_in[3];
  const float* embed   = (const float*)d_in[4];
  const float* edge_W1 = (const float*)d_in[5];
  const float* edge_b1 = (const float*)d_in[6];
  const float* edge_W2 = (const float*)d_in[7];
  const float* edge_b2 = (const float*)d_in[8];
  const float* node_W  = (const float*)d_in[9];
  const float* node_b  = (const float*)d_in[10];
  const float* upd_W1  = (const float*)d_in[11];
  const float* upd_b1  = (const float*)d_in[12];
  const float* upd_W2  = (const float*)d_in[13];
  const float* upd_b2  = (const float*)d_in[14];
  const float* ro_W1   = (const float*)d_in[15];
  const float* ro_b1   = (const float*)d_in[16];
  const float* ro_W2   = (const float*)d_in[17];
  const float* ro_b2   = (const float*)d_in[18];
  float* Eout = (float*)d_out;

  // workspace carve-out (~65 MB total)
  char* wsb = (char*)d_ws;
  size_t off = 0;
  auto alloc = [&](size_t bytes) -> char* {
    char* p = wsb + off;
    off = (off + bytes + 255) & ~(size_t)255;
    return p;
  };
  float* dists = (float*)alloc((size_t)NE_ * 4);
  float* r     = (float*)alloc((size_t)NA_ * NF_ * 4);
  float* h     = (float*)alloc((size_t)NA_ * NF_ * 4);
  float* agg   = (float*)alloc((size_t)NA_ * NF_ * 4);
  float* tN    = (float*)alloc((size_t)NA_ * NF_ * 4);
  _Float16* w1h  = (_Float16*)alloc(3L * NGP_ * NF_ * 2);
  _Float16* w2h  = (_Float16*)alloc(3L * NF_ * NF_ * 2);
  _Float16* nWh  = (_Float16*)alloc(3L * NF_ * NF_ * 2);
  _Float16* u1h  = (_Float16*)alloc(3L * NF_ * NF_ * 2);
  _Float16* u2h  = (_Float16*)alloc(3L * NF_ * NF_ * 2);
  _Float16* roWh = (_Float16*)alloc((size_t)NF_ * 64 * 2);

  k_zero<<<1, 128, 0, stream>>>(Eout, NMOL_);

  // fp32 -> fp16 weight staging
  long np = 3L * NGP_ * NF_;
  k_cvt_pad_w1<<<(unsigned)((np + 255) / 256), 256, 0, stream>>>(edge_W1, w1h);
  long nw = 3L * NF_ * NF_;
  k_cvt<<<(unsigned)((nw + 255) / 256), 256, 0, stream>>>(edge_W2, w2h, nw);
  k_cvt<<<(unsigned)((nw + 255) / 256), 256, 0, stream>>>(node_W,  nWh, nw);
  k_cvt<<<(unsigned)((nw + 255) / 256), 256, 0, stream>>>(upd_W1,  u1h, nw);
  k_cvt<<<(unsigned)((nw + 255) / 256), 256, 0, stream>>>(upd_W2,  u2h, nw);
  long nr = (long)NF_ * 64;
  k_cvt<<<(unsigned)((nr + 255) / 256), 256, 0, stream>>>(ro_W1, roWh, nr);

  k_dist<<<(NE_ + 255) / 256, 256, 0, stream>>>(xyz, nbr, dists, NE_);
  long natf = (long)NA_ * NF_;
  k_embed<<<(unsigned)((natf + 255) / 256), 256, 0, stream>>>(z, embed, r, natf);

  const int gE = (NE_ + 127) / 128;
  const int gN = (NA_ + 127) / 128;
  for (int l = 0; l < NCONV_; ++l) {
    // h = r @ node_W + b
    k_gemm<128, 128, 0><<<gN, 256, 0, stream>>>(
        r, nWh + (size_t)l * NF_ * NF_, node_b + l * NF_, h, NA_);
    // agg = scatter of messages (fused gaussians -> filter MLP -> scatter)
    k_zero<<<(unsigned)((natf + 255) / 256), 256, 0, stream>>>(agg, natf);
    k_edge_fused<<<gE, 256, 0, stream>>>(
        dists, w1h + (size_t)l * NGP_ * NF_, edge_b1 + l * NF_,
        w2h + (size_t)l * NF_ * NF_, edge_b2 + l * NF_, nbr, h, agg, NE_);
    // r += ssp(agg @ U1 + b1) @ U2 + b2   (residual fused via MODE=2)
    k_gemm<128, 128, 1><<<gN, 256, 0, stream>>>(
        agg, u1h + (size_t)l * NF_ * NF_, upd_b1 + l * NF_, tN, NA_);
    k_gemm<128, 128, 2><<<gN, 256, 0, stream>>>(
        tN, u2h + (size_t)l * NF_ * NF_, upd_b2 + l * NF_, r, NA_);
  }

  // readout: e_atom = ssp(r @ ro_W1 + b1) @ ro_W2 + b2 ; segment-sum by mol id
  k_gemm<128, 64, 1><<<gN, 256, 0, stream>>>(r, roWh, ro_b1, tN, NA_);
  k_readout2<<<(NA_ + 255) / 256, 256, 0, stream>>>(tN, ro_W2, ro_b2, mol, Eout, NA_);
}
